// BaselineTransformer_23545010716770
// MI455X (gfx1250) — matrix operand order repaired and checked by reading, not compiled, not run on hardware
//
#include <hip/hip_runtime.h>
#include <hip/hip_bf16.h>

// ---------------------------------------------------------------------------
// CDNA5 (gfx1250) bf16 WMMA transformer forward.
//   VOCAB=32000 D=1024 H=16 DH=64 FF=4096 L=4 S=2048
// ---------------------------------------------------------------------------

typedef __attribute__((ext_vector_type(16))) __bf16        bf16x16;
typedef __attribute__((ext_vector_type(8)))  float         f32x8;
typedef __attribute__((ext_vector_type(4)))  unsigned int  u32x4;
typedef __attribute__((ext_vector_type(4)))  int           i32x4;

union FragU { bf16x16 v; u32x4 q[2]; };

// A-matrix 16x32 bf16 fragment: lane holds row (lane&15); per ISA table the
// lane's 8 VGPRs map to dwords {4h..4h+3, 8+4h..8+4h+3} of the 32-element row
// (h = lane>>4).  `row` points at the 32 contiguous bf16 of this lane's row.
__device__ __forceinline__ bf16x16 load_frag_a(const __bf16* row, int half) {
  FragU u;
  const u32x4* p = (const u32x4*)row;
  u.q[0] = p[half];
  u.q[1] = p[2 + half];
  return u.v;
}

// B-matrix 32x16 bf16 fragment: lane holds column (lane&15); same k order as A:
// dwords {4h..4h+3, 8+4h..8+4h+3} of the 32 contiguous bf16 of this
// lane's column of B (i.e. a row of B^T; we always store B transposed).
__device__ __forceinline__ bf16x16 load_frag_b(const __bf16* col, int half) {
  FragU u;
  const u32x4* p = (const u32x4*)col;
  u.q[0] = p[half];
  u.q[1] = p[2 + half];
  return u.v;
}

// ---- async global -> LDS staging (CDNA5 ASYNCcnt path, guarded) -----------
__device__ __forceinline__ void stage_b128(const __bf16* g, __bf16* l) {
#if __has_builtin(__builtin_amdgcn_global_load_async_to_lds_b128)
  __builtin_amdgcn_global_load_async_to_lds_b128(
      (__attribute__((address_space(1))) i32x4*)((void*)g),
      (__attribute__((address_space(3))) i32x4*)((void*)l), 0, 0);
#else
  *(u32x4*)l = *(const u32x4*)g;
#endif
}
__device__ __forceinline__ void stage_wait() {
#if __has_builtin(__builtin_amdgcn_global_load_async_to_lds_b128)
#if __has_builtin(__builtin_amdgcn_s_wait_asynccnt)
  __builtin_amdgcn_s_wait_asynccnt(0);
#else
  asm volatile("s_wait_asynccnt 0" ::: "memory");
#endif
#endif
}

// ---------------------------------------------------------------------------
// Elementwise / conversion kernels
// ---------------------------------------------------------------------------
__global__ void __launch_bounds__(256)
cvt_f32_bf16_kernel(const float* __restrict__ src, __bf16* __restrict__ dst, long n) {
  long i = (long)blockIdx.x * 256 + threadIdx.x;
  if (i < n) dst[i] = (__bf16)src[i];
}

// src: fp32 [K][N] row-major  ->  dst: bf16 [N][K] row-major (B^T)
__global__ void __launch_bounds__(256)
wtrans_kernel(const float* __restrict__ src, __bf16* __restrict__ dst, int N, int K) {
  __shared__ float t[32][33];
  const int nb = blockIdx.x * 32, kb = blockIdx.y * 32;
  const int tx = threadIdx.x & 31, ty = threadIdx.x >> 5;
#pragma unroll
  for (int i = 0; i < 32; i += 8)
    t[ty + i][tx] = src[(long)(kb + ty + i) * N + nb + tx];
  __syncthreads();
#pragma unroll
  for (int i = 0; i < 32; i += 8)
    dst[(long)(nb + ty + i) * K + kb + tx] = (__bf16)t[tx][ty + i];
}

// V part of qkv (bf16 [S][1024] strided 3072) -> VT bf16 [1024][2048]
__global__ void __launch_bounds__(256)
vtrans_kernel(const __bf16* __restrict__ src, __bf16* __restrict__ dst) {
  __shared__ __bf16 t[32][33];
  const int cb = blockIdx.x * 32;   // (head*64 + d)
  const int sb = blockIdx.y * 32;   // sequence position
  const int tx = threadIdx.x & 31, ty = threadIdx.x >> 5;
#pragma unroll
  for (int i = 0; i < 32; i += 8)
    t[ty + i][tx] = src[(long)(sb + ty + i) * 3072 + cb + tx];
  __syncthreads();
#pragma unroll
  for (int i = 0; i < 32; i += 8)
    dst[(long)(cb + ty + i) * 2048 + sb + tx] = t[tx][ty + i];
}

// x[s][d] = tok_emb[ids[s]][d] + sinusoidal_pe(s, d)
__global__ void __launch_bounds__(256)
embed_kernel(const int* __restrict__ ids, const float* __restrict__ temb,
             float* __restrict__ x) {
  const int s = blockIdx.x;
  const int d = blockIdx.y * 256 + threadIdx.x;
  const int id = ids[s];
  const int even = d & ~1;
  const float ang = (float)s * expf((float)even * (-9.210340371976184f / 1024.f));
  const float pe = (d & 1) ? cosf(ang) : sinf(ang);
  x[(long)s * 1024 + d] = temb[(long)id * 1024 + d] + pe;
}

// One block per row; fp32 in -> bf16 out.  D = 1024 fixed.
__global__ void __launch_bounds__(256)
layernorm_kernel(const float* __restrict__ x, const float* __restrict__ g,
                 const float* __restrict__ b, __bf16* __restrict__ out) {
  __shared__ float red[16];
  const int row = blockIdx.x;
  const float* xr = x + (long)row * 1024;
  const int wave = threadIdx.x >> 5, lane = threadIdx.x & 31;

  float v[4];
  float s = 0.f;
#pragma unroll
  for (int i = 0; i < 4; ++i) { v[i] = xr[threadIdx.x + i * 256]; s += v[i]; }
#pragma unroll
  for (int m = 16; m; m >>= 1) s += __shfl_xor(s, m, 32);
  if (lane == 0) red[wave] = s;
  __syncthreads();
  float tot = 0.f;
#pragma unroll
  for (int i = 0; i < 8; ++i) tot += red[i];
  const float mu = tot * (1.f / 1024.f);

  float s2 = 0.f;
#pragma unroll
  for (int i = 0; i < 4; ++i) { const float d = v[i] - mu; s2 += d * d; }
#pragma unroll
  for (int m = 16; m; m >>= 1) s2 += __shfl_xor(s2, m, 32);
  if (lane == 0) red[8 + wave] = s2;
  __syncthreads();
  float tot2 = 0.f;
#pragma unroll
  for (int i = 0; i < 8; ++i) tot2 += red[8 + i];
  const float rstd = rsqrtf(tot2 * (1.f / 1024.f) + 1e-5f);

#pragma unroll
  for (int i = 0; i < 4; ++i) {
    const int c = threadIdx.x + i * 256;
    out[(long)row * 1024 + c] = (__bf16)((v[i] - mu) * rstd * g[c] + b[c]);
  }
}

// ---------------------------------------------------------------------------
// WMMA GEMM:  C[M][N] = A_bf16[M][K] * (BT_bf16[N][K])^T  + epilogue
//   grid = (N/64, M/128), block = 256 (8 waves); each wave: 16(M) x 64(N).
//   The 64-column B strip is staged into LDS once per block (double-buffered,
//   async global->LDS when available) and shared by all 8 waves.
// ---------------------------------------------------------------------------
__global__ void __launch_bounds__(256)
gemm_wmma_kernel(const __bf16* __restrict__ A, int lda,
                 const __bf16* __restrict__ BT, int ldb,
                 int K,
                 const float* __restrict__ bias,
                 const float* __restrict__ res, int ldr,
                 float* __restrict__ Cf, int ldcf,
                 __bf16* __restrict__ Cb, int ldcb,
                 int gelu) {
  __shared__ __align__(16) __bf16 bsh[2][64][32];   // [buf][B column][k chunk]
  const int lane = threadIdx.x & 31;
  const int wave = threadIdx.x >> 5;
  const int half = lane >> 4;
  const int n16  = lane & 15;
  const long m0 = (long)blockIdx.y * 128 + wave * 16;
  const long n0 = (long)blockIdx.x * 64;

  // staging role of this thread: one b128 (16 bytes) per 64x32 chunk
  const int cr = threadIdx.x >> 2;        // 0..63: row of B^T strip
  const int cs = (threadIdx.x & 3) * 8;   // bf16 offset within 32-elem chunk row
  const __bf16* bsrc = BT + (n0 + cr) * (long)ldb + cs;
  __bf16* bdst[2] = { &bsh[0][cr][cs], &bsh[1][cr][cs] };

  f32x8 acc[4];
#pragma unroll
  for (int t = 0; t < 4; ++t)
#pragma unroll
    for (int v = 0; v < 8; ++v) acc[t][v] = 0.f;

  const __bf16* arow = A + (m0 + n16) * lda;

  // prologue: stage chunk 0
  stage_b128(bsrc, bdst[0]);
  stage_wait();
  __syncthreads();

  int buf = 0;
  for (int k = 0; k < K; k += 32) {
    if (k + 32 < K) stage_b128(bsrc + k + 32, bdst[buf ^ 1]);
    __builtin_prefetch(arow + k + 256, 0, 1);

    const bf16x16 af = load_frag_a(arow + k, half);
#pragma unroll
    for (int t = 0; t < 4; ++t) {
      const bf16x16 bf = load_frag_b(&bsh[buf][t * 16 + n16][0], half);
      acc[t] = __builtin_amdgcn_wmma_f32_16x16x32_bf16(
          false, af, false, bf, (short)0, acc[t], false, false);
    }
    stage_wait();
    __syncthreads();
    buf ^= 1;
  }

#pragma unroll
  for (int t = 0; t < 4; ++t) {
    const long col = n0 + t * 16 + n16;
    const float bv = bias ? bias[col] : 0.f;
#pragma unroll
    for (int v = 0; v < 8; ++v) {
      const long row = m0 + v + 8 * half;
      float val = acc[t][v] + bv;
      if (gelu) val = 0.5f * val * (1.f + erff(val * 0.70710678118f));
      if (res) val += res[row * (long)ldr + col];
      if (Cf) Cf[row * (long)ldcf + col] = val;
      if (Cb) Cb[row * (long)ldcb + col] = (__bf16)val;
    }
  }
}

// ---------------------------------------------------------------------------
// Flash attention: one wave per (head, 16-query tile).  qkv bf16 [S][3072]
// (Q at col 0, K at col 1024), VT bf16 [head*64+d][S].  out bf16 [S][1024].
// ---------------------------------------------------------------------------
__global__ void __launch_bounds__(256)
attention_kernel(const __bf16* __restrict__ qkv, const __bf16* __restrict__ VT,
                 __bf16* __restrict__ out) {
  __shared__ __align__(16) __bf16 psh[8][16][32];   // per-wave P tile
  const int lane = threadIdx.x & 31;
  const int wave = threadIdx.x >> 5;
  const int half = lane >> 4;
  const int n16  = lane & 15;
  const int task = blockIdx.x * 8 + wave;
  const int head = task >> 7;          // / 128 query tiles
  const int q0   = (task & 127) << 4;

  const __bf16* Qb = qkv + head * 64;
  const __bf16* Kb = qkv + 1024 + head * 64;

  const bf16x16 a0 = load_frag_a(Qb + (long)(q0 + n16) * 3072, half);
  const bf16x16 a1 = load_frag_a(Qb + (long)(q0 + n16) * 3072 + 32, half);

  float m[8], l[8];
  f32x8 o[4];
#pragma unroll
  for (int v = 0; v < 8; ++v) { m[v] = -3.0e38f; l[v] = 0.f; }
#pragma unroll
  for (int j = 0; j < 4; ++j)
#pragma unroll
    for (int v = 0; v < 8; ++v) o[j][v] = 0.f;

  const int kend = q0 + 16;
  for (int kb = 0; kb < kend; kb += 32) {
    // ---- scores: two 16x16 tiles covering keys kb..kb+31 -----------------
    f32x8 s0, s1;
#pragma unroll
    for (int v = 0; v < 8; ++v) { s0[v] = 0.f; s1[v] = 0.f; }
    {
      const __bf16* kr0 = Kb + (long)(kb + n16) * 3072;
      const __bf16* kr1 = Kb + (long)(kb + 16 + n16) * 3072;
      s0 = __builtin_amdgcn_wmma_f32_16x16x32_bf16(false, a0, false,
            load_frag_b(kr0, half), (short)0, s0, false, false);
      s0 = __builtin_amdgcn_wmma_f32_16x16x32_bf16(false, a1, false,
            load_frag_b(kr0 + 32, half), (short)0, s0, false, false);
      s1 = __builtin_amdgcn_wmma_f32_16x16x32_bf16(false, a0, false,
            load_frag_b(kr1, half), (short)0, s1, false, false);
      s1 = __builtin_amdgcn_wmma_f32_16x16x32_bf16(false, a1, false,
            load_frag_b(kr1 + 32, half), (short)0, s1, false, false);
    }
    const bool boundary = (kb + 32 > q0);
#pragma unroll
    for (int v = 0; v < 8; ++v) {
      s0[v] *= 0.125f;
      s1[v] *= 0.125f;
      if (boundary) {
        const int qrow = q0 + (half << 3) + v;
        if (kb + n16 > qrow)      s0[v] = -1e9f;
        if (kb + 16 + n16 > qrow) s1[v] = -1e9f;
      }
    }
    // ---- online softmax (rows live across 16-lane half groups) -----------
    float rm[8], rs[8];
#pragma unroll
    for (int v = 0; v < 8; ++v) rm[v] = fmaxf(s0[v], s1[v]);
#pragma unroll
    for (int mk = 1; mk < 16; mk <<= 1)
#pragma unroll
      for (int v = 0; v < 8; ++v) rm[v] = fmaxf(rm[v], __shfl_xor(rm[v], mk, 32));
#pragma unroll
    for (int v = 0; v < 8; ++v) {
      const float mn  = fmaxf(m[v], rm[v]);
      const float scl = expf(m[v] - mn);
      s0[v] = expf(s0[v] - mn);
      s1[v] = expf(s1[v] - mn);
      rs[v] = s0[v] + s1[v];
      m[v]  = mn;
      l[v]  = l[v] * scl;
#pragma unroll
      for (int j = 0; j < 4; ++j) o[j][v] *= scl;
    }
#pragma unroll
    for (int mk = 1; mk < 16; mk <<= 1)
#pragma unroll
      for (int v = 0; v < 8; ++v) rs[v] += __shfl_xor(rs[v], mk, 32);
#pragma unroll
    for (int v = 0; v < 8; ++v) l[v] += rs[v];

    // ---- C-layout P -> LDS -> A-layout fragment --------------------------
#pragma unroll
    for (int v = 0; v < 8; ++v) {
      psh[wave][v + 8 * half][n16]      = (__bf16)s0[v];
      psh[wave][v + 8 * half][16 + n16] = (__bf16)s1[v];
    }
    asm volatile("s_wait_dscnt 0" ::: "memory");
    const bf16x16 pa = load_frag_a(&psh[wave][n16][0], half);

    // ---- O += P * V  (V^T rows are contiguous over keys) -----------------
#pragma unroll
    for (int j = 0; j < 4; ++j) {
      const bf16x16 bv =
          load_frag_b(VT + (long)(head * 64 + j * 16 + n16) * 2048 + kb, half);
      o[j] = __builtin_amdgcn_wmma_f32_16x16x32_bf16(
          false, pa, false, bv, (short)0, o[j], false, false);
    }
  }

#pragma unroll
  for (int v = 0; v < 8; ++v) {
    const float inv = 1.f / l[v];
    const long row = q0 + v + 8 * half;
#pragma unroll
    for (int j = 0; j < 4; ++j)
      out[row * 1024 + head * 64 + j * 16 + n16] = (__bf16)(o[j][v] * inv);
  }
}

// ---------------------------------------------------------------------------
// Host side
// ---------------------------------------------------------------------------
extern "C" void kernel_launch(void* const* d_in, const int* in_sizes, int n_in,
                              void* d_out, int out_size, void* d_ws, size_t ws_size,
                              hipStream_t stream) {
  (void)in_sizes; (void)n_in; (void)out_size; (void)ws_size;
  const long S = 2048, D = 1024, HH = 16, DH = 64, FF = 4096, NL = 4, VOC = 32000;

  const int*   ids    = (const int*)  d_in[0];
  const float* temb   = (const float*)d_in[1];
  const float* qkv_w  = (const float*)d_in[2];
  const float* out_w  = (const float*)d_in[3];
  const float* w1     = (const float*)d_in[4];
  const float* b1     = (const float*)d_in[5];
  const float* w2     = (const float*)d_in[6];
  const float* b2     = (const float*)d_in[7];
  const float* ln1_g  = (const float*)d_in[8];
  const float* ln1_b  = (const float*)d_in[9];
  const float* ln2_g  = (const float*)d_in[10];
  const float* ln2_b  = (const float*)d_in[11];
  const float* lnf_g  = (const float*)d_in[12];
  const float* lnf_b  = (const float*)d_in[13];
  float* logits = (float*)d_out;

  // ---- workspace carve-up (all 256B aligned) -----------------------------
  char* ws = (char*)d_ws;
  size_t off = 0;
  auto carve = [&](size_t bytes) -> void* {
    void* p = ws + off;
    off = (off + bytes + 255) & ~(size_t)255;
    return p;
  };
  __bf16* temb_bf = (__bf16*)carve(VOC * D * 2);
  __bf16* qkvT    = (__bf16*)carve(NL * 3 * D * D * 2);   // [l][3072][1024]
  __bf16* outT    = (__bf16*)carve(NL * D * D * 2);       // [l][1024][1024]
  __bf16* w1T     = (__bf16*)carve(NL * FF * D * 2);      // [l][4096][1024]
  __bf16* w2T     = (__bf16*)carve(NL * D * FF * 2);      // [l][1024][4096]
  float*  x       = (float*) carve(S * D * 4);
  __bf16* h       = (__bf16*)carve(S * D * 2);
  __bf16* qkvb    = (__bf16*)carve(S * 3 * D * 2);        // [S][3072]
  __bf16* VTb     = (__bf16*)carve(HH * DH * S * 2);      // [1024][2048]
  __bf16* attnb   = (__bf16*)carve(S * D * 2);
  __bf16* ffb     = (__bf16*)carve(S * FF * 2);

  // ---- one-time (per call) weight conversion -----------------------------
  cvt_f32_bf16_kernel<<<(int)((VOC * D) / 256), 256, 0, stream>>>(temb, temb_bf, VOC * D);
  for (int l = 0; l < NL; ++l) {
    wtrans_kernel<<<dim3(96, 32),  256, 0, stream>>>(qkv_w + (long)l * D * 3 * D,
                                                     qkvT + (long)l * 3 * D * D, 3072, 1024);
    wtrans_kernel<<<dim3(32, 32),  256, 0, stream>>>(out_w + (long)l * D * D,
                                                     outT + (long)l * D * D, 1024, 1024);
    wtrans_kernel<<<dim3(128, 32), 256, 0, stream>>>(w1 + (long)l * D * FF,
                                                     w1T + (long)l * FF * D, 4096, 1024);
    wtrans_kernel<<<dim3(32, 128), 256, 0, stream>>>(w2 + (long)l * FF * D,
                                                     w2T + (long)l * D * FF, 1024, 4096);
  }

  // ---- embedding ---------------------------------------------------------
  embed_kernel<<<dim3(2048, 4), 256, 0, stream>>>(ids, temb, x);

  // ---- layers ------------------------------------------------------------
  for (int l = 0; l < NL; ++l) {
    const __bf16* qkvT_l = qkvT + (long)l * 3 * D * D;
    const __bf16* outT_l = outT + (long)l * D * D;
    const __bf16* w1T_l  = w1T  + (long)l * FF * D;
    const __bf16* w2T_l  = w2T  + (long)l * D * FF;

    layernorm_kernel<<<2048, 256, 0, stream>>>(x, ln1_g + l * D, ln1_b + l * D, h);

    // qkv = h @ qkv_w   (bf16 out, [S][3072])
    gemm_wmma_kernel<<<dim3(48, 16), 256, 0, stream>>>(
        h, 1024, qkvT_l, 1024, 1024,
        nullptr, nullptr, 0, nullptr, 0, qkvb, 3072, 0);

    vtrans_kernel<<<dim3(32, 64), 256, 0, stream>>>(qkvb + 2048, VTb);
    attention_kernel<<<256, 256, 0, stream>>>(qkvb, VTb, attnb);

    // x += attn @ out_w
    gemm_wmma_kernel<<<dim3(16, 16), 256, 0, stream>>>(
        attnb, 1024, outT_l, 1024, 1024,
        nullptr, x, 1024, x, 1024, nullptr, 0, 0);

    layernorm_kernel<<<2048, 256, 0, stream>>>(x, ln2_g + l * D, ln2_b + l * D, h);

    // ff = gelu(h @ w1 + b1)  (bf16 out, [S][4096])
    gemm_wmma_kernel<<<dim3(64, 16), 256, 0, stream>>>(
        h, 1024, w1T_l, 1024, 1024,
        b1 + (long)l * FF, nullptr, 0, nullptr, 0, ffb, 4096, 1);

    // x += ff @ w2 + b2
    gemm_wmma_kernel<<<dim3(16, 16), 256, 0, stream>>>(
        ffb, 4096, w2T_l, 4096, 4096,
        b2 + (long)l * D, x, 1024, x, 1024, nullptr, 0, 0);
  }

  // ---- final LN + tied-embedding logits ----------------------------------
  layernorm_kernel<<<2048, 256, 0, stream>>>(x, lnf_g, lnf_b, h);
  gemm_wmma_kernel<<<dim3(500, 16), 256, 0, stream>>>(
      h, 1024, temb_bf, 1024, 1024,
      nullptr, nullptr, 0, logits, 32000, nullptr, 0, 0);
}
